// AttentionWithoutEinops_19799799234653
// MI455X (gfx1250) — matrix operand order repaired and checked by reading, not compiled, not run on hardware
//
#include <hip/hip_runtime.h>

// ---------------------------------------------------------------------------
// Causal MHA for MI455X (gfx1250): bf16 WMMA everywhere, f32 accumulate.
// B=4, S=2048, D=1024, N=16 heads, H=64.
// Register-blocked GEMMs: each wave computes 32x64 (8 wmma accumulators),
// so each 32-deep K step issues 2 A-frags + 4 B-frags (12 b128) per 8 wmmas.
// ---------------------------------------------------------------------------

#define DEV_INLINE __device__ __forceinline__

typedef __attribute__((ext_vector_type(16))) __bf16 v16bf;
typedef __attribute__((ext_vector_type(8)))  float  v8f;

static constexpr int Bn = 4, Sn = 2048, Dn = 1024, NHn = 16, HDn = 64;

union FragAB { v16bf v; unsigned u[8]; uint4 q[2]; };
union FragC  { v8f   v; float    f[8]; };

DEV_INLINE unsigned short f2bf_bits(float x) {
  union { float f; unsigned u; } c{x};
  unsigned r = c.u + 0x7FFFu + ((c.u >> 16) & 1u);  // round-to-nearest-even
  return (unsigned short)(r >> 16);
}

DEV_INLINE v8f wmma_bf16(v16bf a, v16bf b, v8f c) {
  return __builtin_amdgcn_wmma_f32_16x16x32_bf16(
      /*neg_a=*/false, a, /*neg_b=*/false, b,
      /*c_mod=*/(short)0, c, /*reuse_a=*/false, /*reuse_b=*/false);
}

// A-matrix 16x32 bf16 fragment (row = lane&15; ISA interleave:
// lanes 0-15 hold K 0-7 & 16-23, lanes 16-31 hold K 8-15 & 24-31).
DEV_INLINE v16bf load_a_frag(const unsigned short* tile, int ldm, int lane) {
  const unsigned* p = (const unsigned*)(tile + (size_t)(lane & 15) * ldm);
  const int hi = (lane & 16) ? 4 : 0;  // +8 bf16 == +4 dwords
  FragAB f;
#pragma unroll
  for (int j = 0; j < 8; ++j) {
    const int k = (j < 4) ? (2 * j) : (2 * j + 8);  // pair start K
    f.u[j] = p[(k >> 1) + hi];
  }
  return f.v;
}

// B-matrix 32x16 bf16 fragment; colbase points at depth 0 of this lane's
// column (lanes 0-15: K 0-7 & 16-23, lanes 16-31: K 8-15 & 24-31).
DEV_INLINE v16bf load_b_frag(const unsigned short* colbase, int lane) {
  const uint4* p = (const uint4*)(colbase + ((lane & 16) ? 8 : 0));
  FragAB f;
  f.q[0] = p[0];
  f.q[1] = p[2];
  return f.v;
}

// ---------------------------------------------------------------------------
// Stage 0: fp32 -> bf16 conversions
// ---------------------------------------------------------------------------
__global__ void cvt_kernel(const float* __restrict__ in,
                           unsigned short* __restrict__ out, int n) {
  int i = blockIdx.x * blockDim.x + threadIdx.x;
  if (i < n) out[i] = f2bf_bits(in[i]);
}

// W_O [N,H,D] -> woT [D, N*H] bf16 (so projection B-operand has contiguous K)
__global__ void wo_transpose_kernel(const float* __restrict__ wo,
                                    unsigned short* __restrict__ woT) {
  int i = blockIdx.x * blockDim.x + threadIdx.x;  // over N*H*D = 1M
  int d = i & (Dn - 1);
  int t = i >> 10;  // n*64+h
  woT[(size_t)d * (NHn * HDn) + t] = f2bf_bits(wo[(size_t)t * Dn + d]);
}

// ---------------------------------------------------------------------------
// Stage 1: QKV projection.  grid = (S/128, 3*N, B), 128 thr.
// Each wave: 32 rows x 64 cols, K = D = 1024.
// q,k stored [B,N,S,H] (q pre-scaled by 1/8);  v stored transposed [B,N,H,S].
// ---------------------------------------------------------------------------
__global__ __launch_bounds__(128) void qkv_kernel(
    const unsigned short* __restrict__ resid,  // [B,S,D] bf16
    const unsigned short* __restrict__ wq,
    const unsigned short* __restrict__ wk,
    const unsigned short* __restrict__ wv,
    unsigned short* __restrict__ qo,
    unsigned short* __restrict__ ko,
    unsigned short* __restrict__ vto) {
  const int lane = threadIdx.x & 31;
  const int wid  = threadIdx.x >> 5;
  const int s0   = (blockIdx.x * 4 + wid) * 32;  // 32 rows per wave
  const int n    = blockIdx.y & 15;
  const int wt   = blockIdx.y >> 4;  // 0=Q 1=K 2=V
  const int b    = blockIdx.z;
  const int c    = lane & 15;

  const unsigned short* w = (wt == 0) ? wq : (wt == 1) ? wk : wv;
  const unsigned short* atile0 = resid + ((size_t)b * Sn + s0) * Dn;
  const unsigned short* atile1 = atile0 + (size_t)16 * Dn;
  const unsigned short* bcol0 = w + (size_t)(n * HDn + 0 * 16 + c) * Dn;
  const unsigned short* bcol1 = w + (size_t)(n * HDn + 1 * 16 + c) * Dn;
  const unsigned short* bcol2 = w + (size_t)(n * HDn + 2 * 16 + c) * Dn;
  const unsigned short* bcol3 = w + (size_t)(n * HDn + 3 * 16 + c) * Dn;

  FragC acc[2][4];
#pragma unroll
  for (int t = 0; t < 2; ++t)
#pragma unroll
    for (int g = 0; g < 4; ++g)
#pragma unroll
      for (int r = 0; r < 8; ++r) acc[t][g].f[r] = 0.f;

  for (int kk = 0; kk < Dn; kk += 32) {
    const v16bf a0 = load_a_frag(atile0 + kk, Dn, lane);
    const v16bf a1 = load_a_frag(atile1 + kk, Dn, lane);
    const v16bf b0 = load_b_frag(bcol0 + kk, lane);
    const v16bf b1 = load_b_frag(bcol1 + kk, lane);
    const v16bf b2 = load_b_frag(bcol2 + kk, lane);
    const v16bf b3 = load_b_frag(bcol3 + kk, lane);
    acc[0][0].v = wmma_bf16(a0, b0, acc[0][0].v);
    acc[1][0].v = wmma_bf16(a1, b0, acc[1][0].v);
    acc[0][1].v = wmma_bf16(a0, b1, acc[0][1].v);
    acc[1][1].v = wmma_bf16(a1, b1, acc[1][1].v);
    acc[0][2].v = wmma_bf16(a0, b2, acc[0][2].v);
    acc[1][2].v = wmma_bf16(a1, b2, acc[1][2].v);
    acc[0][3].v = wmma_bf16(a0, b3, acc[0][3].v);
    acc[1][3].v = wmma_bf16(a1, b3, acc[1][3].v);
  }

  const int rbase = (lane & 16) ? 8 : 0;
  if (wt == 2) {
    // transposed store: fixed column h, 8 consecutive s -> one b128 store
#pragma unroll
    for (int t = 0; t < 2; ++t)
#pragma unroll
      for (int g = 0; g < 4; ++g) {
        unsigned short tb[8];
#pragma unroll
        for (int r = 0; r < 8; ++r) tb[r] = f2bf_bits(acc[t][g].f[r]);
        uint4 pk;
        pk.x = (unsigned)tb[0] | ((unsigned)tb[1] << 16);
        pk.y = (unsigned)tb[2] | ((unsigned)tb[3] << 16);
        pk.z = (unsigned)tb[4] | ((unsigned)tb[5] << 16);
        pk.w = (unsigned)tb[6] | ((unsigned)tb[7] << 16);
        size_t dst = (((size_t)b * NHn + n) * HDn + g * 16 + c) * Sn +
                     s0 + t * 16 + rbase;
        *(uint4*)(vto + dst) = pk;
      }
  } else {
    unsigned short* o = (wt == 0) ? qo : ko;
    const float scale = (wt == 0) ? 0.125f : 1.0f;  // 1/sqrt(64) folded into q
#pragma unroll
    for (int t = 0; t < 2; ++t) {
      size_t base =
          (((size_t)b * NHn + n) * Sn + s0 + t * 16 + rbase) * HDn + c;
#pragma unroll
      for (int g = 0; g < 4; ++g)
#pragma unroll
        for (int r = 0; r < 8; ++r)
          o[base + (size_t)r * HDn + g * 16] = f2bf_bits(acc[t][g].f[r] * scale);
    }
  }
}

// ---------------------------------------------------------------------------
// Stage 2: streaming (flash) causal attention.
// grid = (S/64, N, B), 128 thr; each wave owns one 16-query tile.
// ---------------------------------------------------------------------------
__global__ __launch_bounds__(128) void attn_kernel(
    const unsigned short* __restrict__ qg,   // [B,N,S,H], pre-scaled
    const unsigned short* __restrict__ kg,   // [B,N,S,H]
    const unsigned short* __restrict__ vtg,  // [B,N,H,S]
    unsigned short* __restrict__ aog) {      // [B,N,S,H]
  __shared__ unsigned short P[4][16 * 32];  // per-wave probability tile

  const int lane = threadIdx.x & 31;
  const int wid  = threadIdx.x >> 5;
  const int c    = lane & 15;
  const int hh   = (lane & 16) ? 1 : 0;
  const int q0   = (blockIdx.x * 4 + wid) * 16;
  const size_t bn = (size_t)blockIdx.z * NHn + blockIdx.y;

  const unsigned short* qtile = qg + (bn * Sn + q0) * HDn;
  const v16bf aq0 = load_a_frag(qtile, HDn, lane);       // h 0..31
  const v16bf aq1 = load_a_frag(qtile + 32, HDn, lane);  // h 32..63

  float m[8], l[8];
  FragC o[4];
#pragma unroll
  for (int r = 0; r < 8; ++r) { m[r] = -3.0e38f; l[r] = 0.f; }
#pragma unroll
  for (int g = 0; g < 4; ++g)
#pragma unroll
    for (int r = 0; r < 8; ++r) o[g].f[r] = 0.f;

  unsigned short* Pw = P[wid];

  for (int kb = 0; kb < q0 + 16; kb += 32) {  // 32-wide key blocks, causal
    // Prefetch next key/value block into cache while this one is consumed.
    if (kb + 32 < q0 + 16) {
      __builtin_prefetch(kg + (bn * Sn + kb + 32 + lane) * HDn, 0, 0);
      __builtin_prefetch(vtg + (bn * HDn + lane) * Sn + kb + 32, 0, 0);
      __builtin_prefetch(vtg + (bn * HDn + 32 + lane) * Sn + kb + 32, 0, 0);
    }

    float s0[8], s1[8];
    {  // sub-tile 0: keys kb .. kb+15
      const unsigned short* kbase = kg + (bn * Sn + kb + c) * HDn;
      FragC a;
#pragma unroll
      for (int r = 0; r < 8; ++r) a.f[r] = 0.f;
      a.v = wmma_bf16(aq0, load_b_frag(kbase, lane), a.v);
      a.v = wmma_bf16(aq1, load_b_frag(kbase + 32, lane), a.v);
      const int kcol = kb + c;
#pragma unroll
      for (int r = 0; r < 8; ++r)
        s0[r] = (kcol <= q0 + hh * 8 + r) ? a.f[r] : -3.0e38f;
    }
    if (kb < q0) {  // sub-tile 1: keys kb+16 .. kb+31 (uniform branch)
      const unsigned short* kbase = kg + (bn * Sn + kb + 16 + c) * HDn;
      FragC a;
#pragma unroll
      for (int r = 0; r < 8; ++r) a.f[r] = 0.f;
      a.v = wmma_bf16(aq0, load_b_frag(kbase, lane), a.v);
      a.v = wmma_bf16(aq1, load_b_frag(kbase + 32, lane), a.v);
      const int kcol = kb + 16 + c;
#pragma unroll
      for (int r = 0; r < 8; ++r)
        s1[r] = (kcol <= q0 + hh * 8 + r) ? a.f[r] : -3.0e38f;
    } else {
#pragma unroll
      for (int r = 0; r < 8; ++r) s1[r] = -3.0e38f;
    }

    // online softmax (row stats replicated across the 16 lanes of each half)
#pragma unroll
    for (int r = 0; r < 8; ++r) {
      float v = fmaxf(s0[r], s1[r]);
#pragma unroll
      for (int msk = 1; msk < 16; msk <<= 1)
        v = fmaxf(v, __shfl_xor(v, msk, 32));
      const float mn   = fmaxf(m[r], v);
      const float corr = __expf(m[r] - mn);
      m[r] = mn;
      const float p0 = __expf(s0[r] - mn);
      const float p1 = __expf(s1[r] - mn);
      float rs = p0 + p1;
#pragma unroll
      for (int msk = 1; msk < 16; msk <<= 1)
        rs += __shfl_xor(rs, msk, 32);
      l[r] = l[r] * corr + rs;
#pragma unroll
      for (int g = 0; g < 4; ++g) o[g].f[r] *= corr;
      const int row = r + 8 * hh;
      Pw[row * 32 + c]      = f2bf_bits(p0);
      Pw[row * 32 + 16 + c] = f2bf_bits(p1);
    }

    // P (16x32) @ V-block (32x64): same-wave LDS ops are in-order -> no barrier
    const v16bf ap = load_a_frag(Pw, 32, lane);
#pragma unroll
    for (int g = 0; g < 4; ++g) {
      const unsigned short* vcol = vtg + (bn * HDn + g * 16 + c) * Sn + kb;
      o[g].v = wmma_bf16(ap, load_b_frag(vcol, lane), o[g].v);
    }
  }

#pragma unroll
  for (int r = 0; r < 8; ++r) {
    const float inv = 1.0f / l[r];
#pragma unroll
    for (int g = 0; g < 4; ++g) o[g].f[r] *= inv;
  }
  const size_t obase = (bn * Sn + q0 + 8 * hh) * HDn + c;
#pragma unroll
  for (int g = 0; g < 4; ++g)
#pragma unroll
    for (int r = 0; r < 8; ++r)
      aog[obase + (size_t)r * HDn + g * 16] = f2bf_bits(o[g].f[r]);
}

// ---------------------------------------------------------------------------
// Stage 3: output projection: out[b,s,d] = sum_{n,h} ao[b,n,s,h] * woT[d, n*64+h]
// grid = (S/128, D/64, B), 128 thr.  Each wave: 32 s-rows x 64 d-cols.
// ---------------------------------------------------------------------------
__global__ __launch_bounds__(128) void proj_kernel(
    const unsigned short* __restrict__ ao,   // [B,N,S,H]
    const unsigned short* __restrict__ woT,  // [D, N*H]
    float* __restrict__ out) {               // [B,S,D] fp32
  const int lane = threadIdx.x & 31;
  const int wid  = threadIdx.x >> 5;
  const int c    = lane & 15;
  const int s0   = (blockIdx.x * 4 + wid) * 32;
  const int d0   = blockIdx.y * 64;
  const int b    = blockIdx.z;

  FragC acc[2][4];
#pragma unroll
  for (int t = 0; t < 2; ++t)
#pragma unroll
    for (int g = 0; g < 4; ++g)
#pragma unroll
      for (int r = 0; r < 8; ++r) acc[t][g].f[r] = 0.f;

  const unsigned short* bcol0 = woT + (size_t)(d0 + 0 * 16 + c) * (NHn * HDn);
  const unsigned short* bcol1 = woT + (size_t)(d0 + 1 * 16 + c) * (NHn * HDn);
  const unsigned short* bcol2 = woT + (size_t)(d0 + 2 * 16 + c) * (NHn * HDn);
  const unsigned short* bcol3 = woT + (size_t)(d0 + 3 * 16 + c) * (NHn * HDn);

  for (int kk = 0; kk < NHn * HDn; kk += 32) {
    const int n  = kk >> 6;       // head (32-chunks never straddle heads)
    const int h0 = kk & (HDn - 1);
    const unsigned short* atile =
        ao + (((size_t)b * NHn + n) * Sn + s0) * HDn + h0;
    const v16bf a0 = load_a_frag(atile, HDn, lane);
    const v16bf a1 = load_a_frag(atile + (size_t)16 * HDn, HDn, lane);
    const v16bf b0 = load_b_frag(bcol0 + kk, lane);
    const v16bf b1 = load_b_frag(bcol1 + kk, lane);
    const v16bf b2 = load_b_frag(bcol2 + kk, lane);
    const v16bf b3 = load_b_frag(bcol3 + kk, lane);
    acc[0][0].v = wmma_bf16(a0, b0, acc[0][0].v);
    acc[1][0].v = wmma_bf16(a1, b0, acc[1][0].v);
    acc[0][1].v = wmma_bf16(a0, b1, acc[0][1].v);
    acc[1][1].v = wmma_bf16(a1, b1, acc[1][1].v);
    acc[0][2].v = wmma_bf16(a0, b2, acc[0][2].v);
    acc[1][2].v = wmma_bf16(a1, b2, acc[1][2].v);
    acc[0][3].v = wmma_bf16(a0, b3, acc[0][3].v);
    acc[1][3].v = wmma_bf16(a1, b3, acc[1][3].v);
  }

  const int rbase = (lane & 16) ? 8 : 0;
#pragma unroll
  for (int t = 0; t < 2; ++t) {
    size_t base = ((size_t)b * Sn + s0 + t * 16 + rbase) * Dn + d0 + c;
#pragma unroll
    for (int g = 0; g < 4; ++g)
#pragma unroll
      for (int r = 0; r < 8; ++r)
        out[base + (size_t)r * Dn + g * 16] = acc[t][g].f[r];
  }
}

// ---------------------------------------------------------------------------
extern "C" void kernel_launch(void* const* d_in, const int* in_sizes, int n_in,
                              void* d_out, int out_size, void* d_ws,
                              size_t ws_size, hipStream_t stream) {
  (void)in_sizes; (void)n_in; (void)out_size; (void)ws_size;
  const float* residual = (const float*)d_in[0];
  const float* W_Q = (const float*)d_in[1];
  const float* W_K = (const float*)d_in[2];
  const float* W_V = (const float*)d_in[3];
  const float* W_O = (const float*)d_in[4];
  float* out = (float*)d_out;

  char* ws = (char*)d_ws;
  size_t off = 0;
  auto take = [&](size_t elems) {
    unsigned short* p = (unsigned short*)(ws + off);
    off += elems * sizeof(unsigned short);
    return p;
  };
  const size_t nRes = (size_t)Bn * Sn * Dn;        // 8.4M
  const size_t nW   = (size_t)NHn * HDn * Dn;      // 1M
  const size_t nQKV = (size_t)Bn * NHn * Sn * HDn; // 8.4M
  unsigned short* resid_bf = take(nRes);
  unsigned short* wq_bf    = take(nW);
  unsigned short* wk_bf    = take(nW);
  unsigned short* wv_bf    = take(nW);
  unsigned short* woT_bf   = take(nW);
  unsigned short* q_bf     = take(nQKV);
  unsigned short* k_bf     = take(nQKV);
  unsigned short* vt_bf    = take(nQKV);
  unsigned short* ao_bf    = take(nQKV);

  cvt_kernel<<<(int)((nRes + 255) / 256), 256, 0, stream>>>(residual, resid_bf,
                                                            (int)nRes);
  cvt_kernel<<<(int)((nW + 255) / 256), 256, 0, stream>>>(W_Q, wq_bf, (int)nW);
  cvt_kernel<<<(int)((nW + 255) / 256), 256, 0, stream>>>(W_K, wk_bf, (int)nW);
  cvt_kernel<<<(int)((nW + 255) / 256), 256, 0, stream>>>(W_V, wv_bf, (int)nW);
  wo_transpose_kernel<<<(int)((nW + 255) / 256), 256, 0, stream>>>(W_O, woT_bf);

  qkv_kernel<<<dim3(Sn / 128, 3 * NHn, Bn), 128, 0, stream>>>(
      resid_bf, wq_bf, wk_bf, wv_bf, q_bf, k_bf, vt_bf);
  attn_kernel<<<dim3(Sn / 64, NHn, Bn), 128, 0, stream>>>(q_bf, k_bf, vt_bf,
                                                          ao_bf);
  proj_kernel<<<dim3(Sn / 128, Dn / 64, Bn), 128, 0, stream>>>(ao_bf, woT_bf,
                                                               out);
}
